// MyModel_65609920414091
// MI455X (gfx1250) — matrix operand order repaired and checked by reading, not compiled, not run on hardware
//
#include <hip/hip_runtime.h>
#include <hip/hip_bf16.h>

typedef _Float16 h16;
typedef __attribute__((ext_vector_type(16))) _Float16 v16h;
typedef __attribute__((ext_vector_type(8)))  _Float16 v8h;
typedef __attribute__((ext_vector_type(8)))  float    v8f;

#define FDIM 32
#define USE_ASYNC 1   // CDNA5 GLOBAL_LOAD_ASYNC_TO_LDS_B128 staging path

__device__ __forceinline__ float sigm(float x) { return 1.0f / (1.0f + expf(-x)); }

// ---------------------------------------------------------------------------
// Conv-as-implicit-GEMM with WMMA f16, K tiled by 32, 32x128 output tile.
//   D(32 x 128 per workgroup) = im2col(act)(32 x K) * Bpack(K x 128)
// CI (8 or 32) is compile-time: im2col index math is shifts only, and each
// thread's 8 consecutive K halves are 16 contiguous bytes in memory -> the
// LDS stage is one async global->LDS b128 copy per thread (zeros for OOB).
// RECUR=false : input conv, acc init = bias[n], out = f16 xg buffer
// RECUR=true  : recurrent conv, acc init = xg[b,t,p,n], out = f32 gate buffer
// Block = 128 threads (4 waves); wave w owns N-blocks 2w, 2w+1; 4 WMMA / kb.
// ---------------------------------------------------------------------------
template <bool RECUR, int CI>
__global__ __launch_bounds__(128)
void convgemm_wmma(const h16* __restrict__ act,
                   const h16* __restrict__ bpack,
                   const float* __restrict__ bias,
                   const h16* __restrict__ xg,
                   float* __restrict__ gout,
                   h16* __restrict__ xgout,
                   int Hin, int Win,
                   int Ho, int Wo,
                   int KB, int Ktot, int pad,
                   int T, int t)
{
    __shared__ h16 ldsA[32 * 32];   // 32 M-rows x 32 K halves (2 KB)

    const int tid  = threadIdx.x;
    const int wave = tid >> 5;
    const int lane = tid & 31;
    const int m0   = blockIdx.x << 5;
    const int HoWo = Ho * Wo;

    const int nb0  = wave * 2, nb1 = nb0 + 1;
    const int ncol = lane & 15;
    const int mro  = (lane >> 4) << 3;   // C/D layout: lanes>=16 hold M rows +8

    v8f acc00, acc01, acc10, acc11;      // [mtile][nblock]
#pragma unroll
    for (int r = 0; r < 8; ++r) {
#pragma unroll
        for (int mt = 0; mt < 2; ++mt) {
            const int m = m0 + mt * 16 + r + mro;
            float v0, v1;
            if (RECUR) {
                const int b = m / HoWo;
                const int p = m - b * HoWo;
                const long xi = ((long)((b * T + t) * HoWo + p)) * 128;
                v0 = (float)xg[xi + nb0 * 16 + ncol];
                v1 = (float)xg[xi + nb1 * 16 + ncol];
            } else {
                v0 = bias[nb0 * 16 + ncol];
                v1 = bias[nb1 * 16 + ncol];
            }
            if (mt == 0) { acc00[r] = v0; acc01[r] = v1; }
            else         { acc10[r] = v0; acc11[r] = v1; }
        }
    }

    // staging coords: thread t fills row (t>>2), 8 consecutive k at (t&3)*8
    const int mloc = tid >> 2;
    const int kseg = tid & 3;
    const int ms   = m0 + mloc;
    const int bs   = ms / HoWo;
    const int ps   = ms - bs * HoWo;
    const int ysv  = ps / Wo;
    const int xsv  = ps - ysv * Wo;
    const int ldsidx = (mloc << 5) + (kseg << 3);
#if USE_ASYNC
    const unsigned ldsoff = (unsigned)(unsigned long long)(uintptr_t)&ldsA[ldsidx];
#endif

    // A-fragment gather bases (ISA 16-bit A 16x32 layout)
    const int Ml    = lane & 15;
    const int kbase = (lane & 16) ? 8 : 0;

    for (int kb = 0; kb < KB; ++kb) {
        __syncthreads();
        {
            const int k0  = (kb << 5) + (kseg << 3);
            const int ci0 = k0 & (CI - 1);       // CI is a power of two
            const int rr  = k0 / CI;             // ky*3 + kx
            const int kx  = rr % 3;
            const int ky  = rr / 3;
            const int yy  = ysv + ky - pad;
            const int xx  = xsv + kx - pad;
            const bool valid = (k0 < Ktot) && yy >= 0 && yy < Hin &&
                               xx >= 0 && xx < Win;
            if (valid) {
                const h16* gp = act + (((long)bs * Hin + yy) * Win + xx) * CI + ci0;
#if USE_ASYNC
                asm volatile("global_load_async_to_lds_b128 %0, %1, off"
                             :: "v"(ldsoff),
                                "v"((unsigned long long)(uintptr_t)gp)
                             : "memory");
#else
                *(v8h*)&ldsA[ldsidx] = *(const v8h*)gp;
#endif
            } else {
                v8h z;
#pragma unroll
                for (int j = 0; j < 8; ++j) z[j] = (h16)0.f;
                *(v8h*)&ldsA[ldsidx] = z;
            }
        }
#if USE_ASYNC
        asm volatile("s_wait_asynccnt 0x0" ::: "memory");
#endif
        __syncthreads();

        // A fragments: halves 0..7 -> K=kbase..kbase+7 ; 8..15 -> +16
        v8h a0lo = *(const v8h*)&ldsA[(Ml << 5) + kbase];
        v8h a0hi = *(const v8h*)&ldsA[(Ml << 5) + kbase + 16];
        v8h a1lo = *(const v8h*)&ldsA[((Ml + 16) << 5) + kbase];
        v8h a1hi = *(const v8h*)&ldsA[((Ml + 16) << 5) + kbase + 16];
        v16h a0, a1;
#pragma unroll
        for (int j = 0; j < 8; ++j) {
            a0[j] = a0lo[j]; a0[8 + j] = a0hi[j];
            a1[j] = a1lo[j]; a1[8 + j] = a1hi[j];
        }

        // B fragments: pre-packed contiguous 16 halves per lane
        const h16* bp0 = bpack + (((long)(kb * 8 + nb0) * 32) + lane) * 16;
        const h16* bp1 = bpack + (((long)(kb * 8 + nb1) * 32) + lane) * 16;
        v8h b0lo = *(const v8h*)(bp0);
        v8h b0hi = *(const v8h*)(bp0 + 8);
        v8h b1lo = *(const v8h*)(bp1);
        v8h b1hi = *(const v8h*)(bp1 + 8);
        v16h bf0, bf1;
#pragma unroll
        for (int j = 0; j < 8; ++j) {
            bf0[j] = b0lo[j]; bf0[8 + j] = b0hi[j];
            bf1[j] = b1lo[j]; bf1[8 + j] = b1hi[j];
        }

        acc00 = __builtin_amdgcn_wmma_f32_16x16x32_f16(false, a0, false, bf0,
                                                       (short)0, acc00, false, false);
        acc01 = __builtin_amdgcn_wmma_f32_16x16x32_f16(false, a0, false, bf1,
                                                       (short)0, acc01, false, false);
        acc10 = __builtin_amdgcn_wmma_f32_16x16x32_f16(false, a1, false, bf0,
                                                       (short)0, acc10, false, false);
        acc11 = __builtin_amdgcn_wmma_f32_16x16x32_f16(false, a1, false, bf1,
                                                       (short)0, acc11, false, false);
    }

#pragma unroll
    for (int r = 0; r < 8; ++r) {
#pragma unroll
        for (int mt = 0; mt < 2; ++mt) {
            const int m = m0 + mt * 16 + r + mro;
            const float v0 = (mt == 0) ? acc00[r] : acc10[r];
            const float v1 = (mt == 0) ? acc01[r] : acc11[r];
            if (RECUR) {
                gout[(long)m * 128 + nb0 * 16 + ncol] = v0;
                gout[(long)m * 128 + nb1 * 16 + ncol] = v1;
            } else {
                xgout[(long)m * 128 + nb0 * 16 + ncol] = (h16)v0;
                xgout[(long)m * 128 + nb1 * 16 + ncol] = (h16)v1;
            }
        }
    }
}

// Pack conv weights (3,3,Ci,128) f32 -> f16 B fragments [kb][nb][lane][16]
// lanes 0-15: K = j (j<8), 8+j (j>=8), N = lane ; lanes 16-31: K = 8+j (j<8), 16+j (j>=8), N = lane-16
__global__ void pack_weights(const float* __restrict__ W, h16* __restrict__ bp,
                             int Ci, int Ktot, int total)
{
    int idx = blockIdx.x * blockDim.x + threadIdx.x;
    if (idx >= total) return;
    const int j    = idx & 15;
    const int lane = (idx >> 4) & 31;
    const int nb   = (idx >> 9) & 7;
    const int kb   = idx >> 12;
    const int n    = nb * 16 + (lane & 15);
    const int kl   = j + ((j & 8) ? 8 : 0) + ((lane & 16) ? 8 : 0);
    const int k    = (kb << 5) + kl;
    float v = 0.f;
    if (k < Ktot) {
        const int ci = k % Ci;
        const int rr = k / Ci;                       // ky*3+kx
        v = W[((long)(rr * Ci + ci)) * 128 + n];
    }
    bp[idx] = (h16)v;
}

__global__ void f32_to_f16(const float* __restrict__ in, h16* __restrict__ out, int n)
{
    int i = blockIdx.x * blockDim.x + threadIdx.x;
    if (i < n) out[i] = (h16)in[i];
}

__global__ void f16_to_f32(const h16* __restrict__ in, float* __restrict__ out, int n)
{
    int i = blockIdx.x * blockDim.x + threadIdx.x;
    if (i < n) out[i] = (float)in[i];
}

// LSTM pointwise update: gates (M x 128: i|f|c|o), c f32 state, h f16; also
// writes h into the next-layer input sequence at time t.
__global__ void lstm_update(const float* __restrict__ g, float* __restrict__ c,
                            h16* __restrict__ h, h16* __restrict__ seq,
                            int HoWo, int T, int t, int total)
{
    int i = blockIdx.x * blockDim.x + threadIdx.x;
    if (i >= total) return;
    const int f = i & 31;
    const int m = i >> 5;
    const float* gm = g + (long)m * 128;
    const float gi = gm[f], gf = gm[32 + f], gc = gm[64 + f], go = gm[96 + f];
    float cc = c[i];
    cc = sigm(gf) * cc + sigm(gi) * tanhf(gc);
    const float hh = sigm(go) * tanhf(cc);
    c[i] = cc;
    h[i] = (h16)hh;
    const int b = m / HoWo;
    const int p = m - b * HoWo;
    seq[((long)((b * T + t) * HoWo + p)) * 32 + f] = (h16)hh;
}

__global__ void dense_f32(const float* __restrict__ A, const float* __restrict__ W,
                          const float* __restrict__ bias, float* __restrict__ out,
                          int K, int N, int total, int relu)
{
    int idx = blockIdx.x * blockDim.x + threadIdx.x;
    if (idx >= total) return;
    const int b = idx / N;
    const int n = idx - b * N;
    const float* a = A + (long)b * K;
    float s = bias[n];
    for (int k = 0; k < K; ++k) s += a[k] * W[(long)k * N + n];
    out[idx] = relu ? fmaxf(s, 0.f) : s;
}

__global__ void duel_head(const float* __restrict__ z, const float* __restrict__ WA,
                          const float* __restrict__ bA, const float* __restrict__ WV,
                          const float* __restrict__ bV, float* __restrict__ Q)
{
    int b = threadIdx.x;
    if (b >= 32) return;
    const float* zb = z + b * 256;
    float a0 = bA[0], a1 = bA[1], a2 = bA[2], v = bV[0];
    for (int k = 0; k < 256; ++k) {
        const float zk = zb[k];
        a0 += zk * WA[k * 3 + 0];
        a1 += zk * WA[k * 3 + 1];
        a2 += zk * WA[k * 3 + 2];
        v  += zk * WV[k];
    }
    const float mn = (a0 + a1 + a2) * (1.f / 3.f);
    Q[b * 3 + 0] = v + a0 - mn;
    Q[b * 3 + 1] = v + a1 - mn;
    Q[b * 3 + 2] = v + a2 - mn;
}

extern "C" void kernel_launch(void* const* d_in, const int* in_sizes, int n_in,
                              void* d_out, int out_size, void* d_ws, size_t ws_size,
                              hipStream_t stream)
{
    (void)in_sizes; (void)n_in; (void)out_size; (void)ws_size;
    const float* state = (const float*)d_in[0];
    const float *Wl[5], *Ul[5], *bl[5];
    for (int l = 0; l < 5; ++l) {
        Wl[l] = (const float*)d_in[1 + 3 * l];
        Ul[l] = (const float*)d_in[2 + 3 * l];
        bl[l] = (const float*)d_in[3 + 3 * l];
    }
    const float* Wd1 = (const float*)d_in[16];
    const float* bd1 = (const float*)d_in[17];
    const float* Wd2 = (const float*)d_in[18];
    const float* bd2 = (const float*)d_in[19];
    const float* WA  = (const float*)d_in[20];
    const float* bA  = (const float*)d_in[21];
    const float* WV  = (const float*)d_in[22];
    const float* bV  = (const float*)d_in[23];

    size_t off = 0;
    auto carve = [&](size_t bytes) -> char* {
        char* p = (char*)d_ws + off;
        off += (bytes + 255) & ~(size_t)255;
        return p;
    };
    h16*   seqA = (h16*)carve((size_t)16 << 20);   // activation sequence ping
    h16*   seqB = (h16*)carve((size_t)16 << 20);   // activation sequence pong
    h16*   xg   = (h16*)carve((size_t)40 << 20);   // input-conv gates (f16)
    float* gbuf = (float*)carve((size_t)8 << 20);  // recurrent gates (f32)
    float* cbuf = (float*)carve((size_t)2 << 20);  // LSTM cell state
    h16*   hbuf = (h16*)carve((size_t)1 << 20);    // LSTM hidden state
    h16*   bpW  = (h16*)carve((size_t)128 << 10);  // packed W fragments
    h16*   bpU  = (h16*)carve((size_t)128 << 10);  // packed U fragments
    float* zin  = (float*)carve((size_t)32 * 6272 * sizeof(float));
    float* z1   = (float*)carve((size_t)32 * 256 * sizeof(float));
    float* z2   = (float*)carve((size_t)32 * 256 * sizeof(float));

    const int B = 32, T = 10;

    { // state (B,T,24,24,8) f32 -> f16
        int n = 32 * 10 * 24 * 24 * 8;
        f32_to_f16<<<(n + 255) / 256, 256, 0, stream>>>(state, seqA, n);
    }

    h16* cur = seqA;
    h16* nxt = seqB;
    for (int l = 0; l < 5; ++l) {
        const int Hin = 24 - 2 * l, Win = Hin;
        const int Ci  = (l == 0) ? 8 : FDIM;
        const int Ho  = Hin - 2, Wo = Ho;
        const int HoWo = Ho * Wo;
        const int KtotW = 9 * Ci, KBW = (KtotW + 31) / 32;
        const int KtotU = 9 * FDIM, KBU = 9;

        { int tot = KBW * 8 * 32 * 16;
          pack_weights<<<(tot + 255) / 256, 256, 0, stream>>>(Wl[l], bpW, Ci, KtotW, tot); }
        { int tot = KBU * 8 * 32 * 16;
          pack_weights<<<(tot + 255) / 256, 256, 0, stream>>>(Ul[l], bpU, FDIM, KtotU, tot); }

        hipMemsetAsync(hbuf, 0, (size_t)B * HoWo * FDIM * sizeof(h16), stream);
        hipMemsetAsync(cbuf, 0, (size_t)B * HoWo * FDIM * sizeof(float), stream);

        // time-independent input conv over all B*T images (VALID), bias folded
        {
            const int M2 = B * T * HoWo;   // multiple of 32 (B*T = 320)
            if (l == 0)
                convgemm_wmma<false, 8><<<M2 / 32, 128, 0, stream>>>(
                    cur, bpW, bl[l], nullptr, nullptr, xg,
                    Hin, Win, Ho, Wo, KBW, KtotW, 0, 1, 0);
            else
                convgemm_wmma<false, FDIM><<<M2 / 32, 128, 0, stream>>>(
                    cur, bpW, bl[l], nullptr, nullptr, xg,
                    Hin, Win, Ho, Wo, KBW, KtotW, 0, 1, 0);
        }

        // recurrence
        for (int t = 0; t < T; ++t) {
            const int M = B * HoWo;        // multiple of 32 (B = 32)
            convgemm_wmma<true, FDIM><<<M / 32, 128, 0, stream>>>(
                hbuf, bpU, nullptr, xg, gbuf, nullptr,
                Ho, Wo, Ho, Wo, KBU, KtotU, 1, T, t);
            const int tot = M * FDIM;
            lstm_update<<<(tot + 255) / 256, 256, 0, stream>>>(
                gbuf, cbuf, hbuf, nxt, HoWo, T, t, tot);
        }
        h16* tmp = cur; cur = nxt; nxt = tmp;
    }

    // dueling dense head on last hidden state of layer 4: (32,14,14,32)
    { int n = 32 * 14 * 14 * 32;
      f16_to_f32<<<(n + 255) / 256, 256, 0, stream>>>(hbuf, zin, n); }
    { int tot = 32 * 256;
      dense_f32<<<(tot + 255) / 256, 256, 0, stream>>>(zin, Wd1, bd1, z1, 6272, 256, tot, 1); }
    { int tot = 32 * 256;
      dense_f32<<<(tot + 255) / 256, 256, 0, stream>>>(z1, Wd2, bd2, z2, 256, 256, tot, 1); }
    duel_head<<<1, 32, 0, stream>>>(z2, WA, bA, WV, bV, (float*)d_out);
}
